// Encoder_88304527605854
// MI455X (gfx1250) — matrix operand order repaired and checked by reading, not compiled, not run on hardware
//
#include <hip/hip_runtime.h>

// ---------------- model geometry ----------------
#define DIMM   1024
#define NHEAD  8
#define DHH    128
#define NSEQ   2049            // cls + 2048 tokens
#define NP     2112            // padded seq (2112 = 33*64, B*NP divisible by 128)
#define BATCH  2
#define MROWS  (BATCH*NP)      // 4224 = 33 * 128

typedef __attribute__((ext_vector_type(16))) __bf16      v16bf;
typedef __attribute__((ext_vector_type(8)))  float       v8f;
typedef __attribute__((ext_vector_type(4)))  unsigned int vu4;
typedef __attribute__((ext_vector_type(4)))  unsigned int u32x4;
typedef __attribute__((ext_vector_type(8)))  int          i32x8;
typedef __attribute__((ext_vector_type(4)))  int          i32x4;

union Frag { v16bf v; vu4 u[2]; };

static __device__ __forceinline__ unsigned short f2bf(float f) {
  union { float f; unsigned u; } x; x.f = f;
  unsigned r = x.u + 0x7fffu + ((x.u >> 16) & 1u);   // RTNE
  return (unsigned short)(r >> 16);
}
static __device__ __forceinline__ v8f v8zero() {
  v8f z = {0.f,0.f,0.f,0.f,0.f,0.f,0.f,0.f}; return z;
}
static __device__ __forceinline__ v8f wmma_bf16(const Frag& a, const Frag& b, v8f c) {
  return __builtin_amdgcn_wmma_f32_16x16x32_bf16(false, a.v, false, b.v,
                                                 (short)0, c, false, false);
}

// ---- TDM: async 2D tile load global->LDS (CDNA5 Tensor Data Mover) ----------
// D# group1 is built once per kernel (tile shape + strides + LDS row padding);
// group0 carries the per-tile LDS/global addresses. count=1, type=2 ("image").
static __device__ __forceinline__ i32x8 tdm_group1_2d(int K /*row stride & dim0*/) {
  i32x8 g1;
  // data_size=1(2B) | pad_enable | pad_interval=3 (16 DW = 64B rows)
  // | pad_amount=3 (4 DW = 16B pad)  -> 80B LDS row pitch ( = 40 ushorts)
  g1[0] = (int)((1u << 16) | (1u << 20) | (3u << 22) | (3u << 25));
  g1[1] = (int)(((unsigned)K & 0xffffu) << 16);            // tensor_dim0 lo16
  g1[2] = (int)(((unsigned)K >> 16) & 0xffffu);            // dim0 hi16 | dim1 lo16(=0)
  g1[3] = (int)(((1u << 20) >> 16) | (32u << 16));         // dim1 hi16 | tile_dim0=32
  g1[4] = 128;                                             // tile_dim1=128, tile_dim2=0
  g1[5] = K;                                               // tensor_dim0_stride lo32
  g1[6] = 0;                                               // stride hi16 | dim1_stride lo16
  g1[7] = 0;
  return g1;
}
static __device__ __forceinline__ void tdm_load_2d(unsigned lds_addr,
                                                   const void* gaddr,
                                                   i32x8 g1) {
  unsigned long long ga = (unsigned long long)(size_t)gaddr;
  u32x4 g0;
  g0[0] = 1u;                                              // count=1
  g0[1] = lds_addr;                                        // LDS byte address
  g0[2] = (unsigned)ga;                                    // global addr lo32
  g0[3] = (unsigned)((ga >> 32) & 0x01FFFFFFu) | (2u << 30); // addr[56:32] | type=2
  i32x4 z4 = {0, 0, 0, 0};
#if defined(__clang_major__) && (__clang_major__ >= 23)
  i32x8 z8 = {0, 0, 0, 0, 0, 0, 0, 0};
  __builtin_amdgcn_tensor_load_to_lds(g0, g1, z4, z4, z8, 0);
#else
  __builtin_amdgcn_tensor_load_to_lds(g0, g1, z4, z4, 0);
#endif
}

// ---------------- weight convert + transpose: f32 [K][N] -> bf16 [N][K] ------
__global__ void __launch_bounds__(256)
wcvt_t_kernel(const float* __restrict__ W, unsigned short* __restrict__ Wt,
              int K, int N) {
  __shared__ float s[32][33];
  const int x = threadIdx.x & 31, y = threadIdx.x >> 5;   // 32 x 8
  const int k0 = blockIdx.y * 32, n0 = blockIdx.x * 32;
#pragma unroll
  for (int yy = 0; yy < 32; yy += 8)
    s[y + yy][x] = W[(size_t)(k0 + y + yy) * N + n0 + x];
  __syncthreads();
#pragma unroll
  for (int yy = 0; yy < 32; yy += 8)
    Wt[(size_t)(n0 + y + yy) * K + k0 + x] = f2bf(s[x][y + yy]);
}

// ---------------- pack cls + x into padded fp32/bf16 activations -------------
__global__ void __launch_bounds__(256)
init_x_kernel(const float* __restrict__ x_in, const float* __restrict__ cls,
              float* __restrict__ xf, unsigned short* __restrict__ xb) {
  size_t idx = (size_t)blockIdx.x * 256 + threadIdx.x;   // over MROWS*DIMM
  int d = (int)(idx % DIMM);
  size_t t = idx / DIMM;
  int i = (int)(t % NP);
  int b = (int)(t / NP);
  float v;
  if (i == 0)            v = cls[d];
  else if (i <= 2048)    v = x_in[((size_t)b * 2048 + (i - 1)) * DIMM + d];
  else                   v = 0.f;                        // zero pad (keeps WMMA NaN-free)
  xf[idx] = v;
  xb[idx] = f2bf(v);
}

// ---------------- generic bf16 WMMA GEMM, double-buffered TDM staging --------
// C[M,N] = A[M,K](bf16,row-major) * Bt[N,K](bf16,K-major) (+bias)(+res)(+gelu)
// block: 256 threads = 8 waves; tile 128x128x32; wave tile 32x64 (2x4 wmma).
// Wave 0 drives the Tensor Data Mover: tile k+1 streams into the alternate LDS
// buffer (TENSORcnt) while all 8 waves run WMMA on tile k.
template<bool BIAS, bool RES, bool GELU, bool OUTF, bool OUTB>
__global__ void __launch_bounds__(256)
gemm_bf16_kernel(const unsigned short* __restrict__ A,
                 const unsigned short* __restrict__ Bt,
                 const float* __restrict__ bias,
                 const float* __restrict__ res,
                 float* __restrict__ Cf,
                 unsigned short* __restrict__ Cb,
                 int M, int N, int K) {
  __shared__ __align__(16) unsigned short As[2][128][40];   // 40 = 32 + 4DW hw pad
  __shared__ __align__(16) unsigned short Bs[2][128][40];
  const int tid = threadIdx.x, wave = tid >> 5, lane = tid & 31;
  const int l16 = lane & 15, lhalf = lane >> 4;
  const int wm = wave & 3, wn = wave >> 2;
  const int m0 = blockIdx.y * 128, n0 = blockIdx.x * 128;
  const int nk = K >> 5;

  v8f acc[2][4];
#pragma unroll
  for (int mt = 0; mt < 2; ++mt)
#pragma unroll
    for (int nt = 0; nt < 4; ++nt) acc[mt][nt] = v8zero();

  const i32x8 g1 = tdm_group1_2d(K);
  const unsigned ldsA[2] = {(unsigned)(size_t)&As[0][0][0],
                            (unsigned)(size_t)&As[1][0][0]};
  const unsigned ldsB[2] = {(unsigned)(size_t)&Bs[0][0][0],
                            (unsigned)(size_t)&Bs[1][0][0]};
  const unsigned short* Abase = A + (size_t)m0 * K;
  const unsigned short* Bbase = Bt + (size_t)n0 * K;

  if (wave == 0) {                       // prologue: tile 0 -> buffer 0
    tdm_load_2d(ldsA[0], Abase, g1);
    tdm_load_2d(ldsB[0], Bbase, g1);
  }

  for (int k = 0; k < nk; ++k) {
    const int p = k & 1;
    __syncthreads();                     // everyone done reading buffer 1-p
    if (wave == 0) {
      if (k + 1 < nk) {                  // prefetch tile k+1 -> buffer 1-p
        tdm_load_2d(ldsA[1 - p], Abase + (size_t)(k + 1) * 32, g1);
        tdm_load_2d(ldsB[1 - p], Bbase + (size_t)(k + 1) * 32, g1);
        __builtin_amdgcn_s_wait_tensorcnt(2);   // buffer p landed; k+1 in flight
      } else {
        __builtin_amdgcn_s_wait_tensorcnt(0);
      }
    }
    __syncthreads();                     // buffer p visible to all waves

    const unsigned short (*Ap)[40] = As[p];
    const unsigned short (*Bp)[40] = Bs[p];
    Frag af[2];
#pragma unroll
    for (int mt = 0; mt < 2; ++mt) {
      const unsigned short* ab = &Ap[wm * 32 + mt * 16 + l16][0];
      const int kh = lhalf * 16;                // A frag: K = kh..kh+16 contiguous
      af[mt].u[0] = *(const vu4*)(ab + kh);
      af[mt].u[1] = *(const vu4*)(ab + kh + 8);
    }
#pragma unroll
    for (int nt = 0; nt < 4; ++nt) {
      Frag bfg;                                 // B frag: lane=col, K contiguous
      const unsigned short* bb = &Bp[wn * 64 + nt * 16 + l16][lhalf * 16];
      bfg.u[0] = *(const vu4*)(bb);
      bfg.u[1] = *(const vu4*)(bb + 8);
#pragma unroll
      for (int mt = 0; mt < 2; ++mt)
        acc[mt][nt] = wmma_bf16(af[mt], bfg, acc[mt][nt]);
    }
  }

#pragma unroll
  for (int mt = 0; mt < 2; ++mt)
#pragma unroll
    for (int nt = 0; nt < 4; ++nt) {
      const int col = n0 + wn * 64 + nt * 16 + l16;
      float bv = 0.f;
      if constexpr (BIAS) bv = bias[col];
#pragma unroll
      for (int r = 0; r < 8; ++r) {
        const int row = m0 + wm * 32 + mt * 16 + r + 8 * lhalf;
        float v = acc[mt][nt][r] + bv;
        if constexpr (RES)  v += res[(size_t)row * N + col];
        if constexpr (GELU) v = 0.5f * v * (1.0f + erff(v * 0.70710678118654752f));
        if constexpr (OUTF) Cf[(size_t)row * N + col] = v;
        if constexpr (OUTB) Cb[(size_t)row * N + col] = f2bf(v);
      }
    }
}

// ---------------- transpose V per (b,h): qkv -> Vt[B][H][DH][NP] -------------
__global__ void __launch_bounds__(128)
vtrans_kernel(const unsigned short* __restrict__ qkv,
              unsigned short* __restrict__ Vt) {
  const int b = blockIdx.z, h = blockIdx.y, j0 = blockIdx.x * 32;
  const int d = threadIdx.x;                              // 0..127
  unsigned short tmp[32];
#pragma unroll
  for (int jj = 0; jj < 32; ++jj)
    tmp[jj] = qkv[(size_t)(b * NP + j0 + jj) * (3 * DIMM) + 2 * DIMM + h * DHH + d];
  unsigned short* dst = Vt + ((size_t)(b * NHEAD + h) * DHH + d) * NP + j0;
#pragma unroll
  for (int jj = 0; jj < 32; ++jj) dst[jj] = tmp[jj];
}

// ---------------- flash attention (causal), bf16 WMMA ------------------------
// block: 128 threads = 4 waves; 64 query rows per block (16 per wave)
__global__ void __launch_bounds__(128)
attn_kernel(const unsigned short* __restrict__ qkv,   // [B][NP][3*DIMM]
            const unsigned short* __restrict__ Vt,    // [B][H][DH][NP]
            unsigned short* __restrict__ out) {       // [B][NP][DIMM]
  const int b = blockIdx.z, h = blockIdx.y;
  const int row0 = blockIdx.x * 64;
  const int tid = threadIdx.x, wave = tid >> 5, lane = tid & 31;
  const int l16 = lane & 15, lhalf = lane >> 4;
  const float scale = 0.03125f;                         // 1024^-0.5 (reference uses DIM)
  const float NEG = -1e30f;

  __shared__ __align__(16) unsigned short Ks[32][DHH + 8];  // [j][d], stride 272B
  __shared__ __align__(16) unsigned short Vs[DHH][40];      // [d][j], stride 80B
  __shared__ __align__(16) unsigned short Ps[4][16][40];    // per-wave P staging

  // Q fragments: rows = 16 tokens of this wave, 4 K-slices of 32 over d
  Frag qf[4];
  {
    const int tok = row0 + wave * 16 + l16;
    const unsigned short* qb =
        qkv + (size_t)(b * NP + tok) * (3 * DIMM) + h * DHH;
    const int kh = lhalf * 16;
#pragma unroll
    for (int s = 0; s < 4; ++s) {
      qf[s].u[0] = *(const vu4*)(qb + s * 32 + kh);
      qf[s].u[1] = *(const vu4*)(qb + s * 32 + kh + 8);
    }
  }

  v8f oacc[8];
  float mrow[8], lrow[8];
#pragma unroll
  for (int dt = 0; dt < 8; ++dt) oacc[dt] = v8zero();
#pragma unroll
  for (int r = 0; r < 8; ++r) { mrow[r] = NEG; lrow[r] = 0.f; }

  const int jend = (row0 + 64 < NSEQ) ? (row0 + 64) : NSEQ;

  for (int j0 = 0; j0 < jend; j0 += 32) {
    __syncthreads();
    // stage K tile [32][128] row-major (already "Bt" layout for Q*K^T)
    { const int jj = tid >> 2, part = tid & 3;
      const unsigned short* s =
          qkv + (size_t)(b * NP + j0 + jj) * (3 * DIMM) + DIMM + h * DHH + part * 32;
      vu4* d = (vu4*)&Ks[jj][part * 32];
      d[0] = ((const vu4*)s)[0]; d[1] = ((const vu4*)s)[1];
      d[2] = ((const vu4*)s)[2]; d[3] = ((const vu4*)s)[3]; }
    // stage Vt tile [128][32]
    { const unsigned short* s =
          Vt + ((size_t)(b * NHEAD + h) * DHH + tid) * NP + j0;
      vu4* d = (vu4*)&Vs[tid][0];
      d[0] = ((const vu4*)s)[0]; d[1] = ((const vu4*)s)[1];
      d[2] = ((const vu4*)s)[2]; d[3] = ((const vu4*)s)[3]; }
    __syncthreads();

    // S = Q * K^T : two 16-col tiles, 4 d-slices each
    v8f s0 = v8zero(), s1 = v8zero();
#pragma unroll
    for (int ds = 0; ds < 4; ++ds) {
      Frag kf0, kf1;
      const unsigned short* k0p = &Ks[l16][ds * 32 + lhalf * 16];
      const unsigned short* k1p = &Ks[16 + l16][ds * 32 + lhalf * 16];
      kf0.u[0] = *(const vu4*)k0p; kf0.u[1] = *(const vu4*)(k0p + 8);
      kf1.u[0] = *(const vu4*)k1p; kf1.u[1] = *(const vu4*)(k1p + 8);
      s0 = wmma_bf16(qf[ds], kf0, s0);
      s1 = wmma_bf16(qf[ds], kf1, s1);
    }

    // online softmax per row; write P (bf16) into per-wave LDS staging
    float alpha[8];
#pragma unroll
    for (int r = 0; r < 8; ++r) {
      const int tok = row0 + wave * 16 + r + 8 * lhalf;
      const int c0 = j0 + l16, c1 = j0 + 16 + l16;
      float v0 = ((c0 <= tok) && (c0 < NSEQ)) ? s0[r] * scale : NEG;
      float v1 = ((c1 <= tok) && (c1 < NSEQ)) ? s1[r] * scale : NEG;
      float mx = fmaxf(v0, v1);
#pragma unroll
      for (int off = 1; off < 16; off <<= 1)
        mx = fmaxf(mx, __shfl_xor(mx, off, 32));
      const float mn = fmaxf(mrow[r], mx);
      const float a = __expf(mrow[r] - mn);
      const float p0 = __expf(v0 - mn);
      const float p1 = __expf(v1 - mn);
      float rs = p0 + p1;
#pragma unroll
      for (int off = 1; off < 16; off <<= 1)
        rs += __shfl_xor(rs, off, 32);
      mrow[r] = mn;
      lrow[r] = lrow[r] * a + rs;
      alpha[r] = a;
      Ps[wave][r + 8 * lhalf][l16]      = f2bf(p0);
      Ps[wave][r + 8 * lhalf][16 + l16] = f2bf(p1);
    }
#pragma unroll
    for (int dt = 0; dt < 8; ++dt)
#pragma unroll
      for (int r = 0; r < 8; ++r) oacc[dt][r] *= alpha[r];

    // cross-lane LDS dependency: wait for P stores before fragment loads
    asm volatile("s_wait_dscnt 0" ::: "memory");

    Frag pf;
    { const unsigned short* pb = &Ps[wave][l16][0];
      const int kh = lhalf * 16;
      pf.u[0] = *(const vu4*)(pb + kh);
      pf.u[1] = *(const vu4*)(pb + kh + 8); }

    // O += P * V  (8 d-tiles of 16)
#pragma unroll
    for (int dt = 0; dt < 8; ++dt) {
      Frag vf;
      const unsigned short* vb = &Vs[dt * 16 + l16][lhalf * 16];
      vf.u[0] = *(const vu4*)vb;
      vf.u[1] = *(const vu4*)(vb + 8);
      oacc[dt] = wmma_bf16(pf, vf, oacc[dt]);
    }
  }

  float linv[8];
#pragma unroll
  for (int r = 0; r < 8; ++r) linv[r] = (lrow[r] > 0.f) ? (1.0f / lrow[r]) : 0.f;
#pragma unroll
  for (int dt = 0; dt < 8; ++dt)
#pragma unroll
    for (int r = 0; r < 8; ++r) {
      const int tok = row0 + wave * 16 + r + 8 * lhalf;
      out[(size_t)(b * NP + tok) * DIMM + h * DHH + dt * 16 + l16] =
          f2bf(oacc[dt][r] * linv[r]);
    }
}

// ---------------- output copies ----------------------------------------------
__global__ void __launch_bounds__(256)
cls_copy_kernel(const float* __restrict__ xf, float* __restrict__ outc) {
  const int idx = blockIdx.x * 256 + threadIdx.x;     // 2*1024
  const int d = idx & 1023, b = idx >> 10;
  outc[idx] = xf[(size_t)(b * NP) * DIMM + d];
}
__global__ void __launch_bounds__(256)
final_copy_kernel(const float* __restrict__ xf, float* __restrict__ out) {
  size_t idx = (size_t)blockIdx.x * 256 + threadIdx.x; // 2*2049*1024
  const int d = (int)(idx % DIMM);
  size_t t = idx / DIMM;
  const int i = (int)(t % NSEQ);
  const int b = (int)(t / NSEQ);
  out[idx] = xf[((size_t)(b * NP) + i) * DIMM + d];
}

// ---------------- host driver ------------------------------------------------
extern "C" void kernel_launch(void* const* d_in, const int* in_sizes, int n_in,
                              void* d_out, int out_size, void* d_ws, size_t ws_size,
                              hipStream_t stream) {
  (void)in_sizes; (void)n_in; (void)out_size; (void)ws_size;
  const float* x_in = (const float*)d_in[0];
  const float* cls  = (const float*)d_in[1];
  const float* Wqkv[2] = {(const float*)d_in[2],  (const float*)d_in[9]};
  const float* Wo[2]   = {(const float*)d_in[3],  (const float*)d_in[10]};
  const float* bo[2]   = {(const float*)d_in[4],  (const float*)d_in[11]};
  const float* W1[2]   = {(const float*)d_in[5],  (const float*)d_in[12]};
  const float* b1[2]   = {(const float*)d_in[6],  (const float*)d_in[13]};
  const float* W2[2]   = {(const float*)d_in[7],  (const float*)d_in[14]};
  const float* b2[2]   = {(const float*)d_in[8],  (const float*)d_in[15]};

  const size_t SZ_QKV = (size_t)DIMM * 3 * DIMM;
  const size_t SZ_O   = (size_t)DIMM * DIMM;
  const size_t SZ_1   = (size_t)DIMM * 4 * DIMM;
  const size_t SZ_2   = (size_t)4 * DIMM * DIMM;

  char* ws = (char*)d_ws;
  size_t off = 0;
  auto take = [&](size_t bytes) -> char* {
    char* p = ws + off;
    off = (off + bytes + 255) & ~(size_t)255;
    return p;
  };
  unsigned short* WqkvT = (unsigned short*)take(8 * SZ_QKV * 2);
  unsigned short* WoT   = (unsigned short*)take(8 * SZ_O * 2);
  unsigned short* W1T   = (unsigned short*)take(8 * SZ_1 * 2);
  unsigned short* W2T   = (unsigned short*)take(8 * SZ_2 * 2);
  float*          xf    = (float*)take((size_t)MROWS * DIMM * 4);
  unsigned short* xb    = (unsigned short*)take((size_t)MROWS * DIMM * 2);
  unsigned short* qkvb  = (unsigned short*)take((size_t)MROWS * 3 * DIMM * 2);
  unsigned short* Vt    = (unsigned short*)take((size_t)BATCH * NHEAD * DHH * NP * 2);
  unsigned short* attnb = (unsigned short*)take((size_t)MROWS * DIMM * 2);
  unsigned short* hb    = (unsigned short*)take((size_t)MROWS * 4 * DIMM * 2);

  // one-time: convert + transpose all weights to bf16 K-major
  for (int l = 0; l < 8; ++l) {
    const int g = l >> 2, li = l & 3;
    wcvt_t_kernel<<<dim3(3 * DIMM / 32, DIMM / 32), 256, 0, stream>>>(
        Wqkv[g] + (size_t)li * SZ_QKV, WqkvT + (size_t)l * SZ_QKV, DIMM, 3 * DIMM);
    wcvt_t_kernel<<<dim3(DIMM / 32, DIMM / 32), 256, 0, stream>>>(
        Wo[g] + (size_t)li * SZ_O, WoT + (size_t)l * SZ_O, DIMM, DIMM);
    wcvt_t_kernel<<<dim3(4 * DIMM / 32, DIMM / 32), 256, 0, stream>>>(
        W1[g] + (size_t)li * SZ_1, W1T + (size_t)l * SZ_1, DIMM, 4 * DIMM);
    wcvt_t_kernel<<<dim3(DIMM / 32, 4 * DIMM / 32), 256, 0, stream>>>(
        W2[g] + (size_t)li * SZ_2, W2T + (size_t)l * SZ_2, 4 * DIMM, DIMM);
  }

  init_x_kernel<<<dim3((MROWS * DIMM) / 256), 256, 0, stream>>>(x_in, cls, xf, xb);

  const dim3 gQKV(3 * DIMM / 128, MROWS / 128);   // 24 x 33
  const dim3 gO  (DIMM / 128,     MROWS / 128);   //  8 x 33
  const dim3 gW1 (4 * DIMM / 128, MROWS / 128);   // 32 x 33
  for (int l = 0; l < 8; ++l) {
    const int g = l >> 2, li = l & 3;
    // qkv = x @ Wqkv
    gemm_bf16_kernel<false, false, false, false, true>
        <<<gQKV, 256, 0, stream>>>(xb, WqkvT + (size_t)l * SZ_QKV,
                                   nullptr, nullptr, nullptr, qkvb,
                                   MROWS, 3 * DIMM, DIMM);
    vtrans_kernel<<<dim3(NP / 32, NHEAD, BATCH), 128, 0, stream>>>(qkvb, Vt);
    attn_kernel<<<dim3(NP / 64, NHEAD, BATCH), 128, 0, stream>>>(qkvb, Vt, attnb);
    // x = x + attn @ Wo + bo
    gemm_bf16_kernel<true, true, false, true, true>
        <<<gO, 256, 0, stream>>>(attnb, WoT + (size_t)l * SZ_O,
                                 bo[g] + (size_t)li * DIMM, xf, xf, xb,
                                 MROWS, DIMM, DIMM);
    // h = gelu(x @ W1 + b1)
    gemm_bf16_kernel<true, false, true, false, true>
        <<<gW1, 256, 0, stream>>>(xb, W1T + (size_t)l * SZ_1,
                                  b1[g] + (size_t)li * 4 * DIMM,
                                  nullptr, nullptr, hb,
                                  MROWS, 4 * DIMM, DIMM);
    // x = x + h @ W2 + b2
    gemm_bf16_kernel<true, true, false, true, true>
        <<<gO, 256, 0, stream>>>(hb, W2T + (size_t)l * SZ_2,
                                 b2[g] + (size_t)li * DIMM, xf, xf, xb,
                                 MROWS, DIMM, 4 * DIMM);
    if (l == 3)  // snapshot cls tokens after the head stack
      cls_copy_kernel<<<dim3((BATCH * DIMM) / 256), 256, 0, stream>>>(
          xf, (float*)d_out + (size_t)BATCH * NSEQ * DIMM);
  }

  final_copy_kernel<<<dim3((BATCH * NSEQ * DIMM) / 256), 256, 0, stream>>>(
      xf, (float*)d_out);
}
